// Attention_19000935317650
// MI455X (gfx1250) — matrix operand order repaired and checked by reading, not compiled, not run on hardware
//
#include <hip/hip_runtime.h>
#include <cstdint>
#include <cstddef>

// ---------------------------------------------------------------------------
// Flash-attention forward for B=4, LQ=LS=4096, H=1024, fp32 in/out.
// bf16 WMMA (v_wmma_f32_16x16x32_bf16) for both QK^T and PV GEMMs,
// online softmax, logits never touch global memory.
// ---------------------------------------------------------------------------

typedef __attribute__((ext_vector_type(16))) __bf16        v16bf;
typedef __attribute__((ext_vector_type(2)))  __bf16        v2bf;
typedef __attribute__((ext_vector_type(8)))  float         v8f;
typedef __attribute__((ext_vector_type(4)))  unsigned int  v4u;
typedef __attribute__((ext_vector_type(2)))  unsigned int  v2u;
typedef __attribute__((ext_vector_type(4)))  float         v4f;
typedef __attribute__((ext_vector_type(2)))  float         v2f;

constexpr int BATCH = 4;
constexpr int LQ    = 4096;
constexpr int LS    = 4096;
constexpr int H     = 1024;
constexpr int BM    = 32;    // query rows per workgroup
constexpr int BN    = 32;    // source rows per inner iteration
constexpr float SCALE = 0.03125f; // 1/sqrt(1024)

// dynamic LDS layout (bytes)
constexpr size_t SMEM_BYTES =
    (size_t)(BM * H + BN * H + H * BN + BM * BN) * 2 +         // q, kv, vT, p (bf16)
    (size_t)(4 * 256 + BM * 2 + BM * 2 + BM + BM + BM) * 4;    // spart, pmax, psum, m, a, l

union Frag { v16bf v; v4u u[2]; };

// Packed f32x2 -> bf16x2 via generic vector conversion: lets the backend pick
// the native bf16 converter (V_CVT_PK_BF16_F32) if gfx1250 exposes one;
// legalizes to a software RNE sequence otherwise. Compile-safe either way.
__device__ __forceinline__ unsigned int pkbf(float x, float y) {
  v2f f; f.x = x; f.y = y;
  v2bf r = __builtin_convertvector(f, v2bf);
  return __builtin_bit_cast(unsigned int, r);
}

// Two 16B LDS reads -> one 32B bf16 WMMA operand (maps to ds_load_b128 x2)
__device__ __forceinline__ v16bf ld_frag(const unsigned short* p0,
                                         const unsigned short* p1) {
  Frag f;
  f.u[0] = *(const v4u*)p0;
  f.u[1] = *(const v4u*)p1;
  return f.v;
}

__global__ __launch_bounds__(256) void attn_fwd_kernel(
    const float* __restrict__ Q, const float* __restrict__ S,
    float* __restrict__ O) {
  extern __shared__ char smem[];
  unsigned short* qs   = (unsigned short*)smem;       // [BM][H]   Q tile, bf16
  unsigned short* kvs  = qs  + BM * H;                // [BN][H]   K/V tile, row-major
  unsigned short* vts  = kvs + BN * H;                // [H][BN]   V tile, transposed
  unsigned short* ps   = vts + (size_t)H * BN;        // [BM][BN]  P tile, bf16
  float* spart = (float*)(ps + BM * BN);              // [4][16][16] QK split-k partials
  float* pmax  = spart + 4 * 256;                     // [BM][2]
  float* psum  = pmax  + BM * 2;                      // [BM][2]
  float* mrow  = psum  + BM * 2;                      // [BM] running max
  float* arow  = mrow  + BM;                          // [BM] rescale factor
  float* lrow  = arow  + BM;                          // [BM] running sum

  const int tid  = threadIdx.x;
  const int ln   = tid & 31;
  const int w    = tid >> 5;        // wave id 0..7
  const int half = ln >> 4;         // lane 0-15 vs 16-31
  const int l15  = ln & 15;

  const int b  = blockIdx.x / (LQ / BM);
  const int q0 = (blockIdx.x % (LQ / BM)) * BM;

  const float* Qb = Q + ((size_t)b * LQ + q0) * H;
  const float* Sb = S + (size_t)b * LS * H;

  // ---- stage Q tile into LDS as bf16 (once) ----
  for (int it = 0; it < (BM * H / 4) / 256; ++it) {
    int idx = tid + it * 256;           // float4 index
    int row = idx >> 8;                 // H/4 = 256 float4 per row
    int hc  = (idx & 255) * 4;
    v4f f = *(const v4f*)(Qb + (size_t)row * H + hc);
    v2u pk; pk.x = pkbf(f.x, f.y); pk.y = pkbf(f.z, f.w);
    *(v2u*)&qs[row * H + hc] = pk;
  }
  if (tid < BM) { mrow[tid] = -3.0e38f; lrow[tid] = 0.0f; }

  // output accumulators: wave w owns H columns [w*128, w*128+128)
  v8f acc[2][8];
  for (int i = 0; i < 2; ++i)
    for (int jj = 0; jj < 8; ++jj)
      for (int e = 0; e < 8; ++e) acc[i][jj][e] = 0.0f;

  const int kh = w >> 2;          // split-k half for QK^T
  const int mi = (w >> 1) & 1;    // S-tile row block
  const int ni = w & 1;           // S-tile col block

  for (int j = 0; j < LS; j += BN) {
    __syncthreads();  // previous PV readers done before overwriting kv/vT

    // ---- stage K/V tile (bf16, row-major + transposed copy) ----
    const float* Kb = Sb + (size_t)j * H;
    #pragma unroll
    for (int it = 0; it < (BN * H / 4) / 256; ++it) {
      int idx  = tid + it * 256;
      int srow = idx >> 8;
      int hc   = (idx & 255) * 4;
      v4f f = *(const v4f*)(Kb + (size_t)srow * H + hc);
      unsigned int p01 = pkbf(f.x, f.y);
      unsigned int p23 = pkbf(f.z, f.w);
      v2u pk; pk.x = p01; pk.y = p23;
      *(v2u*)&kvs[srow * H + hc] = pk;
      vts[(hc + 0) * BN + srow] = (unsigned short)(p01 & 0xFFFFu);
      vts[(hc + 1) * BN + srow] = (unsigned short)(p01 >> 16);
      vts[(hc + 2) * BN + srow] = (unsigned short)(p23 & 0xFFFFu);
      vts[(hc + 3) * BN + srow] = (unsigned short)(p23 >> 16);
      if (j + BN < LS)  // prefetch next tile -> global_prefetch_b8
        __builtin_prefetch(Kb + (size_t)BN * H + (size_t)srow * H + hc, 0, 1);
    }
    __syncthreads();

    // ---- S-tile = Q @ K^T (split-k over 2 halves of H) ----
    v8f sacc;
    for (int e = 0; e < 8; ++e) sacc[e] = 0.0f;
    {
      const unsigned short* qrow = qs  + (size_t)(mi * 16 + l15) * H;
      const unsigned short* krow = kvs + (size_t)(ni * 16 + l15) * H;
      #pragma unroll
      for (int kk = 0; kk < 16; ++kk) {
        int k0 = kh * 512 + kk * 32;
        // A: 16x32 bf16 layout -> lane<16: K 0-7 & 16-23, lane>=16: +8
        v16bf a  = ld_frag(qrow + k0 + half * 8, qrow + k0 + 16 + half * 8);
        // B: 32x16 bf16 layout -> lane<16: K 0-7 & 16-23, lane>=16: +8
        v16bf bb = ld_frag(krow + k0 + half * 8, krow + k0 + 16 + half * 8);
        sacc = __builtin_amdgcn_wmma_f32_16x16x32_bf16(
            false, a, false, bb, (short)0, sacc, false, false);
      }
    }
    if (w >= 4) {  // second k-half waves publish partials
      float* dst = spart + (w - 4) * 256;
      #pragma unroll
      for (int e = 0; e < 8; ++e) dst[(e + 8 * half) * 16 + l15] = sacc[e];
    }
    __syncthreads();

    // ---- online softmax: row max ----
    if (w < 4) {
      const float* srcp = spart + w * 256;
      #pragma unroll
      for (int e = 0; e < 8; ++e) {
        sacc[e] = (sacc[e] + srcp[(e + 8 * half) * 16 + l15]) * SCALE;
        float x = sacc[e];
        for (int off = 1; off < 16; off <<= 1)
          x = fmaxf(x, __shfl_xor(x, off, 32));
        if (l15 == 0) pmax[(mi * 16 + e + 8 * half) * 2 + ni] = x;
      }
    }
    __syncthreads();
    if (tid < BM) {
      float tm = fmaxf(pmax[tid * 2], pmax[tid * 2 + 1]);
      float mo = mrow[tid];
      float mn = fmaxf(mo, tm);
      arow[tid] = __expf(mo - mn);   // 0 on first tile (mo = -inf)
      mrow[tid] = mn;
    }
    __syncthreads();

    // rescale output accumulators by alpha (all waves)
    #pragma unroll
    for (int mi2 = 0; mi2 < 2; ++mi2)
      #pragma unroll
      for (int e = 0; e < 8; ++e) {
        float al = arow[mi2 * 16 + e + 8 * half];
        #pragma unroll
        for (int ni2 = 0; ni2 < 8; ++ni2) acc[mi2][ni2][e] *= al;
      }

    // exponentiate, row-sum, write bf16 P tile
    if (w < 4) {
      #pragma unroll
      for (int e = 0; e < 8; ++e) {
        int row = mi * 16 + e + 8 * half;
        float p = __expf(sacc[e] - mrow[row]);
        ps[row * BN + ni * 16 + l15] = (unsigned short)(pkbf(p, p) & 0xFFFFu);
        float x = p;
        for (int off = 1; off < 16; off <<= 1)
          x += __shfl_xor(x, off, 32);
        if (l15 == 0) psum[row * 2 + ni] = x;
      }
    }
    __syncthreads();
    if (tid < BM)
      lrow[tid] = lrow[tid] * arow[tid] + psum[tid * 2] + psum[tid * 2 + 1];

    // ---- O += P @ V  (K = BN = 32 -> one WMMA per tile) ----
    {
      const int hbase = w * 128;
      #pragma unroll
      for (int mi2 = 0; mi2 < 2; ++mi2) {
        const unsigned short* prow = ps + (mi2 * 16 + l15) * BN;
        v16bf a = ld_frag(prow + half * 8, prow + 16 + half * 8);
        #pragma unroll
        for (int ni2 = 0; ni2 < 8; ++ni2) {
          const unsigned short* vrow = vts + (size_t)(hbase + ni2 * 16 + l15) * BN;
          v16bf bb = ld_frag(vrow + half * 8, vrow + 16 + half * 8);
          acc[mi2][ni2] = __builtin_amdgcn_wmma_f32_16x16x32_bf16(
              false, a, false, bb, (short)0, acc[mi2][ni2], false, false);
        }
      }
    }
  }
  __syncthreads();

  // ---- epilogue: O = acc / l ----
  float* Ob = O + ((size_t)b * LQ + q0) * H;
  #pragma unroll
  for (int mi2 = 0; mi2 < 2; ++mi2)
    #pragma unroll
    for (int e = 0; e < 8; ++e) {
      int row = mi2 * 16 + e + 8 * half;
      float inv = 1.0f / lrow[row];
      #pragma unroll
      for (int ni2 = 0; ni2 < 8; ++ni2) {
        int col = w * 128 + ni2 * 16 + l15;
        Ob[(size_t)row * H + col] = acc[mi2][ni2][e] * inv;
      }
    }
}

extern "C" void kernel_launch(void* const* d_in, const int* in_sizes, int n_in,
                              void* d_out, int out_size, void* d_ws, size_t ws_size,
                              hipStream_t stream) {
  (void)in_sizes; (void)n_in; (void)out_size; (void)d_ws; (void)ws_size;
  const float* q = (const float*)d_in[0];
  const float* s = (const float*)d_in[1];
  float* o = (float*)d_out;
  dim3 grid(BATCH * (LQ / BM));   // 512 workgroups
  dim3 block(256);                // 8 wave32s
  attn_fwd_kernel<<<grid, block, SMEM_BYTES, stream>>>(q, s, o);
}
